// MultiHeadAttention_57397942943786
// MI455X (gfx1250) — matrix operand order repaired and checked by reading, not compiled, not run on hardware
//
#include <hip/hip_runtime.h>

typedef __bf16 bf16_t;
typedef __attribute__((ext_vector_type(8)))  __bf16 v8bf;
typedef __attribute__((ext_vector_type(16))) __bf16 v16bf;
typedef __attribute__((ext_vector_type(8)))  float  v8f;
typedef __attribute__((ext_vector_type(4)))  unsigned int v4u;
typedef __attribute__((ext_vector_type(8)))  int v8i;
typedef __attribute__((ext_vector_type(4)))  int v4i;

#define E_DIM 1024
#define NH    16
#define HD    64
#define BATCH 4
#define SEQ   1024
#define NTOK  (BATCH * SEQ)        // 4096 tokens
#define PLANE (NTOK * E_DIM)       // 4 Mi elements per plane
#define WSZ   (E_DIM * E_DIM)      // one weight matrix

__device__ __forceinline__ bf16_t to_bf(float f) { return (bf16_t)f; }

__device__ __forceinline__ v16bf combine8(v8bf lo, v8bf hi) {
  v16bf r;
#pragma unroll
  for (int i = 0; i < 8; ++i) { r[i] = lo[i]; r[8 + i] = hi[i]; }
  return r;
}

// --- gfx1250 async global->LDS copy (ASYNCcnt-tracked) ----------------------
__device__ __forceinline__ void async_g2l_b64(unsigned lds_off, const void* base,
                                              int byte_off) {
  asm volatile("global_load_async_to_lds_b64 %0, %1, %2 offset:0"
               :: "v"(lds_off), "v"(byte_off), "s"(base) : "memory");
}
__device__ __forceinline__ void wait_async_le1() {
  asm volatile("s_wait_asynccnt 0x1" ::: "memory");
}
__device__ __forceinline__ void wait_async_le0() {
  asm volatile("s_wait_asynccnt 0x0" ::: "memory");
}

// --- gfx1250 Tensor Data Mover: 2D tile load (TENSORcnt-tracked) ------------
// D# per CDNA5 ISA ch.8: group0 = {flags, lds_addr, global_addr, type=2},
// group1 = {data_size, tensor dims, tile dims, dim0 stride}. 2D -> groups 2/3 zero.
// This toolchain exposes the 6-arg builtin (extra int32x8 group before cpol).
__device__ __forceinline__ void tdm_load_2d(unsigned lds_byte, const void* gaddr,
                                            unsigned tile_d0, unsigned tile_d1,
                                            unsigned stride_elems,
                                            unsigned data_size_code)
{
  unsigned long long ga = (unsigned long long)(size_t)gaddr;
  v4u g0;
  g0[0] = 1u;                                            // count=1 (valid user D#)
  g0[1] = lds_byte;                                      // lds_addr
  g0[2] = (unsigned)(ga & 0xffffffffu);                  // global_addr[31:0]
  g0[3] = (unsigned)((ga >> 32) & 0x01ffffffu) | (2u << 30);  // addr[56:32] | type=2
  const unsigned td0 = 0x7fffffffu, td1 = 0x7fffffffu;   // tensor dims: no OOB clip
  v8i g1;
  g1[0] = (int)(data_size_code << 16);                   // wg_mask=0, data_size
  g1[1] = (int)((td0 & 0xffffu) << 16);                  // atomic_addr=0 | td0[15:0]
  g1[2] = (int)((td0 >> 16) | ((td1 & 0xffffu) << 16));  // td0[31:16] | td1[15:0]
  g1[3] = (int)((td1 >> 16) | (tile_d0 << 16));          // td1[31:16] | tile_dim0
  g1[4] = (int)(tile_d1 & 0xffffu);                      // tile_dim1 | tile_dim2=0
  g1[5] = (int)stride_elems;                             // dim0_stride[31:0]
  g1[6] = 0;                                             // stride hi / dim1_stride
  g1[7] = 0;
  v4i z4 = {0, 0, 0, 0};
  v8i z8 = {0, 0, 0, 0, 0, 0, 0, 0};
  __builtin_amdgcn_tensor_load_to_lds(g0, g1, z4, z4, z8, 0);
}

// ---------------------------------------------------------------------------
// Kernel 0: one-shot transpose of the four weights into bf16 [z][n][k].
// ---------------------------------------------------------------------------
__global__ __launch_bounds__(256)
void transpose_w_kernel(const float* __restrict__ Wq, const float* __restrict__ Wk,
                        const float* __restrict__ Wv, const float* __restrict__ Wo,
                        bf16_t* __restrict__ wt)
{
  __shared__ float tile[32][33];
  const int z = blockIdx.z;
  const float* W = (z == 0) ? Wq : (z == 1) ? Wk : (z == 2) ? Wv : Wo;
  const int kx = blockIdx.x * 32;
  const int ny = blockIdx.y * 32;
  const int tx = threadIdx.x, ty = threadIdx.y;
#pragma unroll
  for (int r = 0; r < 4; ++r)
    tile[ty + 8 * r][tx] = W[(kx + ty + 8 * r) * E_DIM + ny + tx];
  __syncthreads();
#pragma unroll
  for (int r = 0; r < 4; ++r)
    wt[(size_t)z * WSZ + (size_t)(ny + ty + 8 * r) * E_DIM + kx + tx] =
        to_bf(tile[tx][ty + 8 * r]);
}

// ---------------------------------------------------------------------------
// Kernel 1: Q/K/V projection + bias + rotary.  grid = (NTOK/16, E/256, 3),
// block = 128 (4 waves).  A tile (16x32 f32) arrives via the Tensor Data
// Mover (double-buffered, one descriptor per K-step, wave 0 issues);
// B fragments are contiguous b128 loads from pre-transposed bf16 weights.
// ---------------------------------------------------------------------------
__global__ __launch_bounds__(128)
void qkv_rotary_kernel(const float* __restrict__ qin, const float* __restrict__ kin,
                       const float* __restrict__ vin,
                       const bf16_t* __restrict__ wt,
                       const float* __restrict__ bq, const float* __restrict__ bk,
                       const float* __restrict__ bv,
                       bf16_t* __restrict__ ws)
{
  __shared__ float Xs[2][16][32];   // 2 x 2 KB A tiles (f32), TDM double buffer

  const int tid  = threadIdx.x;
  const int wave = tid >> 5;
  const int lane = tid & 31;
  const int t0   = blockIdx.x * 16;
  const int nb   = blockIdx.y * 256;
  const int z    = blockIdx.z;

  const float* X    = (z == 0) ? qin : (z == 1) ? kin : vin;
  const float* bias = (z == 0) ? bq  : (z == 1) ? bk  : bv;
  const bf16_t* wtz = wt + (size_t)z * WSZ;

  const int g  = lane >> 4;
  const int mA = lane & 15;
  const int nB = lane & 15;

  const unsigned xs_base = (unsigned)(size_t)&Xs[0][0][0];
  const float*   xrow0   = X + (size_t)t0 * E_DIM;

  if (wave == 0)
    tdm_load_2d(xs_base, xrow0, /*tile_d0=*/32, /*tile_d1=*/16,
                /*stride=*/E_DIM, /*data_size=4B*/2u);

  v8f acc[4] = {};

  for (int i = 0; i < E_DIM / 32; ++i) {
    const int kb = i * 32;
    const int p  = i & 1;
    if (wave == 0) {
      if (i + 1 < E_DIM / 32) {
        tdm_load_2d(xs_base + (unsigned)((p ^ 1) * 2048), xrow0 + kb + 32,
                    32, 16, E_DIM, 2u);
        __builtin_amdgcn_s_wait_tensorcnt(1);   // current tile landed
      } else {
        __builtin_amdgcn_s_wait_tensorcnt(0);
      }
    }
    __syncthreads();

    // A fragment (f32 LDS -> bf16): elems 0..7 K=g*8+i, 8..15 K=16+g*8+i
    v16bf a;
#pragma unroll
    for (int q = 0; q < 8; ++q) {
      a[q]     = to_bf(Xs[p][mA][g * 8 + q]);
      a[8 + q] = to_bf(Xs[p][mA][16 + g * 8 + q]);
    }
    // load all 4 B fragments first, then the WMMA chain
    v16bf bf[4];
#pragma unroll
    for (int st = 0; st < 4; ++st) {
      const int col = nb + wave * 64 + st * 16 + nB;
      const bf16_t* wrow = wtz + (size_t)col * E_DIM + kb;
      bf[st] = combine8(*(const v8bf*)&wrow[g * 8],
                        *(const v8bf*)&wrow[16 + g * 8]);
    }
    if (kb + 32 < E_DIM)
      __builtin_prefetch(&wtz[(size_t)(nb + tid) * E_DIM + kb + 32], 0, 3);
#pragma unroll
    for (int st = 0; st < 4; ++st)
      acc[st] = __builtin_amdgcn_wmma_f32_16x16x32_bf16(false, a, false, bf[st],
                                                        (short)0, acc[st],
                                                        false, false);
    __syncthreads();   // all reads of buffer p done before it is re-filled
  }

  // Epilogue: bias + rotary (q,k) + store
#pragma unroll
  for (int st = 0; st < 4; ++st) {
#pragma unroll
    for (int r = 0; r < 8; ++r) {
      int m = r + g * 8;
      int t = t0 + m;
      int e = nb + wave * 64 + st * 16 + nB;
      float v = acc[st][r] + bias[e];
      int bb = t >> 10, s = t & 1023;
      int h = e >> 6, d = e & 63;
      if (z < 2) {
        float partner = __shfl_xor(v, 1, 32);   // pair (2j, 2j+1), same row
        int   j   = d >> 1;
        float div = __expf(-0.28782313662425572f * (float)j);
        float ang = __sinf((float)s * div);
        float ct = __cosf(ang), st2 = __sinf(ang);
        v = (d & 1) ? (v * ct + partner * st2) : (v * ct - partner * st2);
        ws[(size_t)z * PLANE + ((size_t)(bb * NH + h) * SEQ + s) * HD + d] = to_bf(v);
      } else {
        ws[2 * (size_t)PLANE + ((size_t)(bb * NH + h) * HD + d) * SEQ + s] = to_bf(v);
      }
    }
  }
}

// ---------------------------------------------------------------------------
// Kernel 2: attention core. grid = (B*H, S/16), block = 256 (8 waves).
// ---------------------------------------------------------------------------
__global__ __launch_bounds__(256)
void attn_kernel(const bf16_t* __restrict__ ws, bf16_t* __restrict__ ctx,
                 float* __restrict__ probs)
{
  __shared__ float sc[16][SEQ];   // 64 KB of the 320 KB WGP LDS

  const int tid  = threadIdx.x;
  const int wave = tid >> 5;
  const int lane = tid & 31;
  const int bh   = blockIdx.x;
  const int s0   = blockIdx.y * 16;

  const bf16_t* qp  = ws + (size_t)bh * SEQ * HD;
  const bf16_t* kp  = ws + (size_t)PLANE + (size_t)bh * SEQ * HD;
  const bf16_t* vtp = ws + 2 * (size_t)PLANE + (size_t)bh * HD * SEQ; // [D][S]

  const int g  = lane >> 4;
  const int mA = lane & 15;
  const int nB = lane & 15;

  v16bf aq0, aq1;
  {
    const bf16_t* row = qp + (size_t)(s0 + mA) * HD;
    aq0 = combine8(*(const v8bf*)&row[g * 8],      *(const v8bf*)&row[16 + g * 8]);
    aq1 = combine8(*(const v8bf*)&row[32 + g * 8], *(const v8bf*)&row[48 + g * 8]);
  }

  // ---- Phase 1: scores = (q k^T) / sqrt(D) into LDS ----
  for (int j = wave; j < SEQ / 16; j += 8) {
    int t0 = j * 16;
    const bf16_t* krow = kp + (size_t)(t0 + nB) * HD;
    v16bf bk0 = combine8(*(const v8bf*)&krow[g * 8],
                         *(const v8bf*)&krow[16 + g * 8]);
    v16bf bk1 = combine8(*(const v8bf*)&krow[32 + g * 8],
                         *(const v8bf*)&krow[48 + g * 8]);
    v8f c = {};
    c = __builtin_amdgcn_wmma_f32_16x16x32_bf16(false, aq0, false, bk0,
                                                (short)0, c, false, false);
    c = __builtin_amdgcn_wmma_f32_16x16x32_bf16(false, aq1, false, bk1,
                                                (short)0, c, false, false);
#pragma unroll
    for (int r = 0; r < 8; ++r)
      sc[r + g * 8][t0 + nB] = c[r] * 0.125f;
  }
  __syncthreads();

  // ---- Phase 2: row softmax; 16 threads per row (shuffle reductions) ----
  {
    const int row = tid >> 4;
    const int c0  = tid & 15;
    float mx = -3.402823466e38f;
#pragma unroll 8
    for (int i = 0; i < 64; ++i) mx = fmaxf(mx, sc[row][c0 + 16 * i]);
#pragma unroll
    for (int off = 1; off < 16; off <<= 1) mx = fmaxf(mx, __shfl_xor(mx, off, 32));
    float sum = 0.f;
#pragma unroll 8
    for (int i = 0; i < 64; ++i) {
      int col = c0 + 16 * i;
      float ev = __expf(sc[row][col] - mx);
      sc[row][col] = ev;
      sum += ev;
    }
#pragma unroll
    for (int off = 1; off < 16; off <<= 1) sum += __shfl_xor(sum, off, 32);
    float inv = 1.0f / sum;
    float* prow = probs + ((size_t)bh * SEQ + (s0 + row)) * SEQ;
#pragma unroll 8
    for (int i = 0; i < 64; ++i) {
      int col = c0 + 16 * i;
      float p = sc[row][col] * inv;
      sc[row][col] = p;
      prow[col]    = p;
    }
  }
  __syncthreads();

  // ---- Phase 3: ctx = probs @ V; waves 0..3, d-tile = wave*16 ----
  if (wave < 4) {
    const int d0 = wave * 16;
    const bf16_t* vrow = vtp + (size_t)(d0 + nB) * SEQ;
    v8f c = {};
    for (int kb = 0; kb < SEQ; kb += 32) {
      v16bf a;
#pragma unroll
      for (int q = 0; q < 8; ++q) {
        a[q]     = to_bf(sc[mA][kb + g * 8 + q]);
        a[8 + q] = to_bf(sc[mA][kb + 16 + g * 8 + q]);
      }
      v16bf b = combine8(*(const v8bf*)&vrow[kb + g * 8],
                         *(const v8bf*)&vrow[kb + 16 + g * 8]);
      c = __builtin_amdgcn_wmma_f32_16x16x32_bf16(false, a, false, b,
                                                  (short)0, c, false, false);
    }
    const int bb = bh >> 4, h = bh & 15;
#pragma unroll
    for (int r = 0; r < 8; ++r) {
      int s = s0 + r + g * 8;
      int e = h * HD + d0 + nB;
      ctx[((size_t)bb * SEQ + s) * E_DIM + e] = to_bf(c[r]);
    }
  }
}

// ---------------------------------------------------------------------------
// Kernel 3: out = ctx @ Wo + bo.  16x256 tiling; bf16 A tile double-buffered
// via global_load_async_to_lds_b64 (ASYNCcnt).
// ---------------------------------------------------------------------------
__global__ __launch_bounds__(128)
void out_proj_kernel(const bf16_t* __restrict__ ctx, const bf16_t* __restrict__ wt,
                     const float* __restrict__ bo, float* __restrict__ out)
{
  __shared__ bf16_t Xs[2][16][32];   // 2 x 1 KB A tiles (bf16)

  const int tid  = threadIdx.x;
  const int wave = tid >> 5;
  const int lane = tid & 31;
  const int t0   = blockIdx.x * 16;
  const int nb   = blockIdx.y * 256;
  const bf16_t* wto = wt + 3 * (size_t)WSZ;

  const int g  = lane >> 4;
  const int mA = lane & 15;
  const int nB = lane & 15;

  const int cm = tid >> 3;           // row 0..15
  const int ck = (tid & 7) * 4;      // 4 bf16 = 8 bytes per thread
  const unsigned lds0 = (unsigned)(size_t)&Xs[0][cm][ck];
  const int gbase = ((t0 + cm) * E_DIM + ck) * 2;   // byte offset of thread's chunk

  async_g2l_b64(lds0, ctx, gbase);

  v8f acc[4] = {};

  for (int i = 0; i < E_DIM / 32; ++i) {
    const int kb = i * 32;
    const int p  = i & 1;
    if (i + 1 < E_DIM / 32) {
      async_g2l_b64(lds0 + (unsigned)((p ^ 1) * 1024), ctx, gbase + (kb + 32) * 2);
      wait_async_le1();
    } else {
      wait_async_le0();
    }
    __syncthreads();

    v16bf a = combine8(*(const v8bf*)&Xs[p][mA][g * 8],
                       *(const v8bf*)&Xs[p][mA][16 + g * 8]);
    v16bf bf[4];
#pragma unroll
    for (int st = 0; st < 4; ++st) {
      const int col = nb + wave * 64 + st * 16 + nB;
      const bf16_t* wrow = wto + (size_t)col * E_DIM + kb;
      bf[st] = combine8(*(const v8bf*)&wrow[g * 8],
                        *(const v8bf*)&wrow[16 + g * 8]);
    }
    if (kb + 32 < E_DIM)
      __builtin_prefetch(&wto[(size_t)(nb + tid) * E_DIM + kb + 32], 0, 3);
#pragma unroll
    for (int st = 0; st < 4; ++st)
      acc[st] = __builtin_amdgcn_wmma_f32_16x16x32_bf16(false, a, false, bf[st],
                                                        (short)0, acc[st],
                                                        false, false);
    __syncthreads();
  }

#pragma unroll
  for (int st = 0; st < 4; ++st) {
#pragma unroll
    for (int r = 0; r < 8; ++r) {
      int t = t0 + r + g * 8;
      int e = nb + wave * 64 + st * 16 + nB;
      out[(size_t)t * E_DIM + e] = acc[st][r] + bo[e];
    }
  }
}

// ---------------------------------------------------------------------------
extern "C" void kernel_launch(void* const* d_in, const int* in_sizes, int n_in,
                              void* d_out, int out_size, void* d_ws, size_t ws_size,
                              hipStream_t stream) {
  const float* query = (const float*)d_in[0];
  const float* key   = (const float*)d_in[1];
  const float* value = (const float*)d_in[2];
  const float* Wq = (const float*)d_in[3];
  const float* bq = (const float*)d_in[4];
  const float* Wk = (const float*)d_in[5];
  const float* bk = (const float*)d_in[6];
  const float* Wv = (const float*)d_in[7];
  const float* bv = (const float*)d_in[8];
  const float* Wo = (const float*)d_in[9];
  const float* bo = (const float*)d_in[10];

  bf16_t* ws  = (bf16_t*)d_ws;                  // q | k | v^T | ctx | W^T x4
  bf16_t* ctx = ws + 3 * (size_t)PLANE;
  bf16_t* wt  = ws + 4 * (size_t)PLANE;         // 40 MB total workspace
  float*  out   = (float*)d_out;
  float*  probs = out + (size_t)NTOK * E_DIM;

  transpose_w_kernel<<<dim3(E_DIM / 32, E_DIM / 32, 4), dim3(32, 8), 0, stream>>>(
      Wq, Wk, Wv, Wo, wt);

  qkv_rotary_kernel<<<dim3(NTOK / 16, E_DIM / 256, 3), dim3(128), 0, stream>>>(
      query, key, value, wt, bq, bk, bv, ws);

  attn_kernel<<<dim3(BATCH * NH, SEQ / 16), dim3(256), 0, stream>>>(ws, ctx, probs);

  out_proj_kernel<<<dim3(NTOK / 16, E_DIM / 256), dim3(128), 0, stream>>>(
      ctx, wt, bo, out);
}
